// Net_45354854645958
// MI455X (gfx1250) — matrix operand order repaired and checked by reading, not compiled, not run on hardware
//
#include <hip/hip_runtime.h>
#include <hip/hip_bf16.h>

// ---------------------------------------------------------------------------
// 2-layer GCN for MI455X (gfx1250, wave32).
//  GEMMs: v_wmma_f32_16x16x32_bf16, fp32 accum, register-prefetch pipelined,
//         bf16 pairs pre-packed in LDS (conflict-free strides),
//         packed hardware fp32->bf16 conversion (v_cvt_pk_bf16_f32).
//  Edge aggregation: L2-resident f32 atomics + global_prefetch_b8 look-ahead.
// ---------------------------------------------------------------------------

typedef __attribute__((ext_vector_type(16))) __bf16 v16bf;
typedef __attribute__((ext_vector_type(8)))  float  v8f;
typedef __attribute__((ext_vector_type(2)))  float  f32x2;
typedef __attribute__((ext_vector_type(2)))  __bf16 bf16x2;

// pack two fp32 -> {bf16 lo, bf16 hi}: vector fptrunc -> v_cvt_pk_bf16_f32
__device__ __forceinline__ unsigned pack2bf(float lo, float hi) {
  union { bf16x2 b; unsigned u; } w;
  w.b = __builtin_convertvector((f32x2){lo, hi}, bf16x2);
  return w.u;
}

union FragAB { v16bf v; unsigned u[8]; };

// ---------------------------------------------------------------------------
// C[M x NCOLS] = op(A)[M x KDIM] @ B[KDIM x NCOLS]   (op = optional ReLU)
// 256 threads = 8 waves; block computes 128 rows x NCOLS; K tiled by 32.
// ---------------------------------------------------------------------------
template<int KDIM, int NCOLS, int WAVES_M, bool RELU>
__global__ __launch_bounds__(256) void gemm_bf16_kernel(
    const float* __restrict__ A, const float* __restrict__ B,
    float* __restrict__ C, int M)
{
  constexpr int WAVES_N = 8 / WAVES_M;
  constexpr int TM = 128 / WAVES_M;       // rows per wave
  constexpr int TN = NCOLS / WAVES_N;     // cols per wave
  constexpr int TILES_M = TM / 16;
  constexpr int TILES_N = TN / 16;
  constexpr int AST = 18;                 // u32 stride per row: 16 k-pairs + 2 pad
  constexpr int BST = NCOLS + 4;          // u32 stride per k-pair row
  constexpr int NB4 = NCOLS / 4;
  constexpr int BIT = NCOLS / 64;         // B staging iters (2 f4 loads each)

  __shared__ unsigned lA32[128 * AST];    // [row][kpair] packed bf16x2
  __shared__ unsigned lB32[16 * BST];     // [kpair][col] packed bf16x2

  const int tid  = threadIdx.x;
  const int lane = tid & 31;
  const int wid  = tid >> 5;
  const int wm   = wid % WAVES_M;
  const int wn   = wid / WAVES_M;
  const int half = lane >> 4;
  const int l16  = lane & 15;
  const int blockRow = blockIdx.x * 128;

  v8f acc[TILES_M][TILES_N];
#pragma unroll
  for (int i = 0; i < TILES_M; ++i)
#pragma unroll
    for (int j = 0; j < TILES_N; ++j)
      acc[i][j] = (v8f){0.f, 0.f, 0.f, 0.f, 0.f, 0.f, 0.f, 0.f};

  float4 ra[4];          // staged A: 4 float4 per thread (128x32 tile)
  float4 rb[BIT][2];     // staged B: k / k+1 row pairs

  auto loadA = [&](int k0) {
#pragma unroll
    for (int it = 0; it < 4; ++it) {
      int idx = it * 256 + tid;           // float4 slot
      int r = idx >> 3;                   // row in tile
      int s = idx & 7;                    // float4 seg in 32 k
      int grow = blockRow + r;
      float4 f = make_float4(0.f, 0.f, 0.f, 0.f);
      if (grow < M)
        f = *(const float4*)(A + (size_t)grow * KDIM + k0 + s * 4);
      ra[it] = f;
    }
  };
  auto loadB = [&](int k0) {
#pragma unroll
    for (int it = 0; it < BIT; ++it) {
      int idx = it * 256 + tid;
      int p = idx / NB4;                  // k-pair row (0..15)
      int s = idx % NB4;                  // float4 seg of columns
      rb[it][0] = *(const float4*)(B + (size_t)(k0 + 2 * p) * NCOLS + s * 4);
      rb[it][1] = *(const float4*)(B + (size_t)(k0 + 2 * p + 1) * NCOLS + s * 4);
    }
  };
  auto stageLDS = [&]() {
#pragma unroll
    for (int it = 0; it < 4; ++it) {
      int idx = it * 256 + tid;
      int r = idx >> 3;
      int s = idx & 7;
      float4 f = ra[it];
      if (RELU) {
        f.x = fmaxf(f.x, 0.f); f.y = fmaxf(f.y, 0.f);
        f.z = fmaxf(f.z, 0.f); f.w = fmaxf(f.w, 0.f);
      }
      uint2 w = make_uint2(pack2bf(f.x, f.y), pack2bf(f.z, f.w));
      *(uint2*)&lA32[r * AST + s * 2] = w;       // 8B aligned
    }
#pragma unroll
    for (int it = 0; it < BIT; ++it) {
      int idx = it * 256 + tid;
      int p = idx / NB4;
      int s = idx % NB4;
      float4 f0 = rb[it][0], f1 = rb[it][1];
      uint4 w = make_uint4(pack2bf(f0.x, f1.x), pack2bf(f0.y, f1.y),
                           pack2bf(f0.z, f1.z), pack2bf(f0.w, f1.w));
      *(uint4*)&lB32[p * BST + s * 4] = w;       // 16B aligned (BST*4 % 16 == 0)
    }
  };

  loadA(0);
  loadB(0);

  for (int k0 = 0; k0 < KDIM; k0 += 32) {
    stageLDS();
    __syncthreads();

    // issue next tile's global loads; they fly while WMMA runs below
    if (k0 + 32 < KDIM) { loadA(k0 + 32); loadB(k0 + 32); }

    // ---- fragments per ISA 16-bit layouts (pairs pre-packed) ----
    FragAB a[TILES_M];
#pragma unroll
    for (int tm = 0; tm < TILES_M; ++tm) {
      int row = wm * TM + tm * 16 + l16;
      const unsigned* ap = &lA32[row * AST + 4 * half];
#pragma unroll
      for (int v = 0; v < 8; ++v)               // kpair = (v&3) + 8*(v>>2) [+4*half]
        a[tm].u[v] = ap[(v & 3) + 8 * (v >> 2)];
    }
    FragAB b[TILES_N];
#pragma unroll
    for (int tn = 0; tn < TILES_N; ++tn) {
      int col = wn * TN + tn * 16 + l16;
      const unsigned* bp = &lB32[half * 4 * BST + col];
#pragma unroll
      for (int v = 0; v < 8; ++v)               // kpair row = (v&3) + 8*(v>>2) [+4*half]
        b[tn].u[v] = bp[((v & 3) + 8 * (v >> 2)) * BST];
    }
#pragma unroll
    for (int tm = 0; tm < TILES_M; ++tm)
#pragma unroll
      for (int tn = 0; tn < TILES_N; ++tn)
        acc[tm][tn] = __builtin_amdgcn_wmma_f32_16x16x32_bf16(
            false, a[tm].v, false, b[tn].v, (short)0, acc[tm][tn], false, false);

    __syncthreads();
  }

  // ---- store C: D layout -> lane (n, half); vgpr r is M = half*8 + r ----
#pragma unroll
  for (int tm = 0; tm < TILES_M; ++tm) {
#pragma unroll
    for (int tn = 0; tn < TILES_N; ++tn) {
      int col = wn * TN + tn * 16 + l16;
#pragma unroll
      for (int r = 0; r < 8; ++r) {
        int row = blockRow + wm * TM + tm * 16 + half * 8 + r;
        if (row < M)
          C[(size_t)row * NCOLS + col] = acc[tm][tn][r];
      }
    }
  }
}

// ---------------------------------------------------------------------------
// Degree / normalization
// ---------------------------------------------------------------------------
__global__ __launch_bounds__(256) void deg_init_kernel(float* deg, int N) {
  int i = blockIdx.x * 256 + threadIdx.x;
  if (i < N) deg[i] = 1.0f;                 // self-loop
}

__global__ __launch_bounds__(256) void deg_count_kernel(
    const int* __restrict__ dst, float* deg, int E) {
  int i = blockIdx.x * 256 + threadIdx.x;
  if (i < E) atomicAdd(&deg[dst[i]], 1.0f);
}

__global__ __launch_bounds__(256) void rsqrt_kernel(float* deg, int N) {
  int i = blockIdx.x * 256 + threadIdx.x;
  if (i < N) deg[i] = rsqrtf(deg[i]);
}

// out[v][j] = bias[j] + inv[v]^2 * H[v][j]   (self-loop term + bias)
template<int W>
__global__ __launch_bounds__(256) void agg_init_kernel(
    const float* __restrict__ H, const float* __restrict__ inv,
    const float* __restrict__ bias, float* __restrict__ out, int N) {
  int i = blockIdx.x * 256 + threadIdx.x;
  if (i >= N * W) return;
  int v = i / W;
  int j = i - v * W;
  float iv = inv[v];
  out[i] = bias[j] + iv * iv * H[i];
}

// ---------------------------------------------------------------------------
// Edge scatter: one wave per edge.  H / out rows are L2-resident
// (51 MB + 51 MB << 192 MB L2) so gathers + atomics stay on-chip.
// ---------------------------------------------------------------------------
__global__ __launch_bounds__(256) void scatter128_kernel(
    const float* __restrict__ H, const float* __restrict__ inv,
    const int* __restrict__ src, const int* __restrict__ dst,
    float* __restrict__ out, int E) {
  int lane = threadIdx.x & 31;
  int w = threadIdx.x >> 5;
  int e = blockIdx.x * 8 + w;
  if (e >= E) return;
  int ep = e + 8;
  if (ep < E) {                              // look-ahead: global_prefetch_b8
    int sp = src[ep];
    __builtin_prefetch(H + (size_t)sp * 128 + lane * 4, 0, 3);
  }
  int s = src[e], d = dst[e];
  float c = inv[s] * inv[d];
  float4 h = *((const float4*)(H + (size_t)s * 128) + lane);
  float* op = out + (size_t)d * 128 + lane * 4;
  atomicAdd(op + 0, c * h.x);
  atomicAdd(op + 1, c * h.y);
  atomicAdd(op + 2, c * h.z);
  atomicAdd(op + 3, c * h.w);
}

__global__ __launch_bounds__(256) void scatter64_kernel(
    const float* __restrict__ H, const float* __restrict__ inv,
    const int* __restrict__ src, const int* __restrict__ dst,
    float* __restrict__ out, int E) {
  int lane = threadIdx.x & 31;
  int w = threadIdx.x >> 5;
  int e = blockIdx.x * 8 + w;
  if (e >= E) return;
  int ep = e + 8;
  if (ep < E) {
    int sp = src[ep];
    __builtin_prefetch(H + (size_t)sp * 64 + lane * 2, 0, 3);
  }
  int s = src[e], d = dst[e];
  float c = inv[s] * inv[d];
  float2 h = *((const float2*)(H + (size_t)s * 64) + lane);
  float* op = out + (size_t)d * 64 + lane * 2;
  atomicAdd(op + 0, c * h.x);
  atomicAdd(op + 1, c * h.y);
}

// ---------------------------------------------------------------------------
// In-place log-softmax over rows of 64; one wave32 per row, 2 elems/lane.
// ---------------------------------------------------------------------------
__global__ __launch_bounds__(256) void logsoftmax64_kernel(float* out, int N) {
  int lane = threadIdx.x & 31;
  int w = threadIdx.x >> 5;
  int v = blockIdx.x * 8 + w;
  if (v >= N) return;
  float* row = out + (size_t)v * 64;
  float a = row[lane];
  float b = row[lane + 32];
  float m = fmaxf(a, b);
#pragma unroll
  for (int off = 16; off; off >>= 1) m = fmaxf(m, __shfl_xor(m, off, 32));
  float s = __expf(a - m) + __expf(b - m);
#pragma unroll
  for (int off = 16; off; off >>= 1) s += __shfl_xor(s, off, 32);
  float ls = m + __logf(s);
  row[lane] = a - ls;
  row[lane + 32] = b - ls;
}

// ---------------------------------------------------------------------------
extern "C" void kernel_launch(void* const* d_in, const int* in_sizes, int n_in,
                              void* d_out, int out_size, void* d_ws, size_t ws_size,
                              hipStream_t stream) {
  const float* x   = (const float*)d_in[0];
  const int*   src = (const int*)d_in[1];
  const int*   dst = (const int*)d_in[2];
  const float* W1  = (const float*)d_in[3];
  const float* b1  = (const float*)d_in[4];
  const float* W2  = (const float*)d_in[5];
  const float* b2  = (const float*)d_in[6];

  const int F_IN = 512, HID = 128, CLS = 64;
  const int N = in_sizes[0] / F_IN;
  const int E = in_sizes[1];

  // workspace layout: H1 | AGG1 | H2 | deg   (~128.4 MB)
  float* H1   = (float*)d_ws;                 // N x 128  (x @ W1)
  float* AGG1 = H1   + (size_t)N * HID;       // N x 128  (layer-1 output)
  float* H2   = AGG1 + (size_t)N * HID;       // N x 64   (relu(AGG1) @ W2)
  float* deg  = H2   + (size_t)N * CLS;       // N        (deg -> inv_sqrt)
  float* out  = (float*)d_out;                // N x 64

  dim3 blk(256);
  int gN  = (N + 255) / 256;
  int gE  = (E + 255) / 256;
  int gM  = (N + 127) / 128;         // GEMM row blocks
  int gEw = (E + 7) / 8;             // one wave per edge
  int gNw = (N + 7) / 8;             // one wave per node row

  // normalization: deg = indegree + 1 ; inv = rsqrt(deg)
  deg_init_kernel<<<gN, blk, 0, stream>>>(deg, N);
  deg_count_kernel<<<gE, blk, 0, stream>>>(dst, deg, E);
  rsqrt_kernel<<<gN, blk, 0, stream>>>(deg, N);

  // layer 1
  gemm_bf16_kernel<512, 128, 4, false><<<gM, blk, 0, stream>>>(x, W1, H1, N);
  agg_init_kernel<128><<<(unsigned)(((size_t)N * 128 + 255) / 256), blk, 0, stream>>>(
      H1, deg, b1, AGG1, N);
  scatter128_kernel<<<gEw, blk, 0, stream>>>(H1, deg, src, dst, AGG1, E);

  // layer 2 (ReLU fused into A-operand bf16 conversion)
  gemm_bf16_kernel<128, 64, 8, true><<<gM, blk, 0, stream>>>(AGG1, W2, H2, N);
  agg_init_kernel<64><<<(unsigned)(((size_t)N * 64 + 255) / 256), blk, 0, stream>>>(
      H2, deg, b2, out, N);
  scatter64_kernel<<<gEw, blk, 0, stream>>>(H2, deg, src, dst, out, E);

  logsoftmax64_kernel<<<gNw, blk, 0, stream>>>(out, N);
}
